// FusedLlamaLowBitMultiDecoderlayer_26895085207995
// MI455X (gfx1250) — compile-verified, matrix operand order checked by reading, not run on hardware
//
#include <hip/hip_runtime.h>

// ---------------------------------------------------------------------------
// Types
// ---------------------------------------------------------------------------
typedef __bf16 bf16_t;
typedef __attribute__((ext_vector_type(16))) __bf16 v16bf;
typedef __attribute__((ext_vector_type(8)))  __bf16 v8bf;
typedef __attribute__((ext_vector_type(8)))  float  v8f;

#define S_LEN 1024
#define HID   2048
#define INTER_ 8192
#define KVW   512     // N_KV_HEADS * HEAD_DIM
#define NH    32
#define NKV   8
#define HD    64

// ---------------------------------------------------------------------------
// Generic bf16 WMMA GEMM:  C[M,N] = A[M,K] * B[N,K]^T  (+ optional residual)
// 256 threads = 8 waves (4 M x 2 N). Wave tile 32 x (BNT/2).
// K staged in 64-deep double-buffered LDS tiles; L2 prefetch 2 tiles ahead.
// Batched over grid.z with element-offset strides (GQA via bDiv on B).
// ---------------------------------------------------------------------------
#define BM  128
#define BK_ 64
#define LDS_STRIDE 72   // 64 + 8 pad (keeps 16B alignment, spreads banks)

__device__ inline v16bf load_frag(const bf16_t* __restrict__ tile, int lane) {
    // 16-bit A/B fragment per CDNA5 ISA: lane&15 selects row (M for A, N for B),
    // lane>=16 selects K-halves {8..15, 24..31}; vector elems 0..7 = K kb..kb+7,
    // elems 8..15 = K kb+16..kb+23.
    int m  = lane & 15;
    int kb = (lane >> 4) << 3;
    const bf16_t* p = tile + m * LDS_STRIDE + kb;
    union { v16bf v; v8bf h[2]; } u;
    u.h[0] = *(const v8bf*)(p);
    u.h[1] = *(const v8bf*)(p + 16);
    return u.v;
}

template <int BNT>
__global__ __launch_bounds__(256)
void gemm_bf16_kernel(const bf16_t* __restrict__ A, long long aBatch, int lda,
                      const bf16_t* __restrict__ B, long long bBatch, int bDiv, int ldb,
                      float* __restrict__ C, bf16_t* __restrict__ Cb,
                      long long cBatch, int ldc,
                      const float* __restrict__ R, int ldr,
                      int K) {
    constexpr int WNF = BNT / 32;             // N-frags per wave (4 or 2)
    constexpr int ACH = BM * (BK_ / 8) / 256; // A 16B-chunks per thread = 4
    constexpr int BCH = BNT * (BK_ / 8) / 256;

    __shared__ __align__(16) bf16_t As[2][BM  * LDS_STRIDE];
    __shared__ __align__(16) bf16_t Bs[2][BNT * LDS_STRIDE];

    const int z = blockIdx.z;
    A += (size_t)z * (size_t)aBatch;
    B += (size_t)(z / bDiv) * (size_t)bBatch;
    if (C)  C  += (size_t)z * (size_t)cBatch;
    if (Cb) Cb += (size_t)z * (size_t)cBatch;

    const int m0 = blockIdx.y * BM;
    const int n0 = blockIdx.x * BNT;
    const int t    = threadIdx.x;
    const int wave = t >> 5;
    const int lane = t & 31;
    const int wm = (wave >> 1) * 32;          // 0,32,64,96
    const int wn = (wave & 1) * (BNT / 2);    // 0 or BNT/2

    v8f acc[2][WNF] = {};

    auto stage = [&](int kk, int buf) {
#pragma unroll
        for (int i = 0; i < ACH; ++i) {
            int c   = t + i * 256;
            int row = c >> 3;                 // 8 chunks per 64-wide row
            int col = (c & 7) << 3;
            *(v8bf*)&As[buf][row * LDS_STRIDE + col] =
                *(const v8bf*)&A[(size_t)(m0 + row) * lda + kk + col];
        }
#pragma unroll
        for (int i = 0; i < BCH; ++i) {
            int c   = t + i * 256;
            int row = c >> 3;
            int col = (c & 7) << 3;
            *(v8bf*)&Bs[buf][row * LDS_STRIDE + col] =
                *(const v8bf*)&B[(size_t)(n0 + row) * ldb + kk + col];
        }
    };

    auto compute = [&](int buf) {
#pragma unroll
        for (int ks = 0; ks < BK_; ks += 32) {
            v16bf a0 = load_frag(&As[buf][(wm     ) * LDS_STRIDE + ks], lane);
            v16bf a1 = load_frag(&As[buf][(wm + 16) * LDS_STRIDE + ks], lane);
#pragma unroll
            for (int f = 0; f < WNF; ++f) {
                v16bf bb = load_frag(&Bs[buf][(wn + f * 16) * LDS_STRIDE + ks], lane);
                acc[0][f] = __builtin_amdgcn_wmma_f32_16x16x32_bf16(
                    false, a0, false, bb, (short)0, acc[0][f], false, false);
                acc[1][f] = __builtin_amdgcn_wmma_f32_16x16x32_bf16(
                    false, a1, false, bb, (short)0, acc[1][f], false, false);
            }
        }
    };

    stage(0, 0);
    __syncthreads();
    int cur = 0;
    for (int kk = 0; kk < K; kk += BK_) {
        if (kk + BK_ < K) stage(kk + BK_, cur ^ 1);
        // L2 prefetch two tiles ahead: one 128B row per thread
        int kk2 = kk + 2 * BK_;
        if (kk2 < K) {
            if (t < BM)
                __builtin_prefetch(&A[(size_t)(m0 + t) * lda + kk2], 0, 3);
            else if (t - BM < BNT)
                __builtin_prefetch(&B[(size_t)(n0 + t - BM) * ldb + kk2], 0, 3);
        }
        compute(cur);
        __syncthreads();
        cur ^= 1;
    }

    // C/D layout: lane&15 = N, lane>=16 adds 8 to M, VGPR r = M row within 8
    const int nIn   = lane & 15;
    const int mHalf = (lane >> 4) << 3;
#pragma unroll
    for (int fm = 0; fm < 2; ++fm)
#pragma unroll
        for (int fn = 0; fn < WNF; ++fn)
#pragma unroll
            for (int r = 0; r < 8; ++r) {
                int m = m0 + wm + fm * 16 + mHalf + r;
                int n = n0 + wn + fn * 16 + nIn;
                float v = acc[fm][fn][r];
                if (R) v += R[(size_t)m * ldr + n];
                if (Cb) Cb[(size_t)m * ldc + n] = (bf16_t)v;
                else    C [(size_t)m * ldc + n] = v;
            }
}

// ---------------------------------------------------------------------------
// Elementwise / reduction kernels
// ---------------------------------------------------------------------------
__global__ void f32_to_bf16_kernel(const float* __restrict__ in,
                                   bf16_t* __restrict__ out, long long n) {
    long long i = (long long)blockIdx.x * blockDim.x + threadIdx.x;
    long long stride = (long long)gridDim.x * blockDim.x;
    for (; i < n; i += stride) out[i] = (bf16_t)in[i];
}

__global__ __launch_bounds__(256)
void rmsnorm_kernel(const float* __restrict__ x, const float* __restrict__ w,
                    bf16_t* __restrict__ out, int H) {
    const int row = blockIdx.x;
    const float* xr = x + (size_t)row * H;
    bf16_t* outr = out + (size_t)row * H;
    float ss = 0.f;
    for (int j = threadIdx.x; j < H; j += 256) { float v = xr[j]; ss += v * v; }
    __shared__ float red[256];
    red[threadIdx.x] = ss; __syncthreads();
    for (int s = 128; s > 0; s >>= 1) {
        if (threadIdx.x < s) red[threadIdx.x] += red[threadIdx.x + s];
        __syncthreads();
    }
    float inv = rsqrtf(red[0] / (float)H + 1e-5f);
    for (int j = threadIdx.x; j < H; j += 256)
        outr[j] = (bf16_t)(xr[j] * inv * w[j]);
}

// RoPE on q: [S, 32*64] fp32 -> bf16 (pos == s for this harness)
__global__ void rope_q_kernel(const float* __restrict__ q,
                              const float* __restrict__ cosc,
                              const float* __restrict__ sinc,
                              bf16_t* __restrict__ q_bf) {
    int idx = blockIdx.x * blockDim.x + threadIdx.x;
    if (idx >= S_LEN * HID) return;
    int s = idx >> 11;
    int d = idx & 63;
    float c  = cosc[s * HD + d];
    float sn = sinc[s * HD + d];
    float v  = q[idx];
    float other = (d < 32) ? -q[idx + 32] : q[idx - 32];
    q_bf[idx] = (bf16_t)(v * c + other * sn);
}

// RoPE on k + pack v: writes k_bf [S,512], vT_bf [8][64][S], new_k/new_v fp32
__global__ void rope_kv_kernel(const float* __restrict__ kf,
                               const float* __restrict__ vf,
                               const float* __restrict__ cosc,
                               const float* __restrict__ sinc,
                               bf16_t* __restrict__ k_bf,
                               bf16_t* __restrict__ vT_bf,
                               float* __restrict__ outK,
                               float* __restrict__ outV) {
    int idx = blockIdx.x * blockDim.x + threadIdx.x;
    if (idx >= S_LEN * KVW) return;
    int s   = idx >> 9;
    int col = idx & 511;
    int kv  = col >> 6;
    int d   = col & 63;
    float c  = cosc[s * HD + d];
    float sn = sinc[s * HD + d];
    float kvv = kf[idx];
    float other = (d < 32) ? -kf[idx + 32] : kf[idx - 32];
    float kr = kvv * c + other * sn;
    k_bf[idx] = (bf16_t)kr;
    size_t o = ((size_t)kv * S_LEN + s) * HD + d;   // [n_kv, S, D]
    outK[o] = kr;
    float vv = vf[idx];
    outV[o] = vv;
    vT_bf[((size_t)kv * HD + d) * S_LEN + s] = (bf16_t)vv;
}

__global__ __launch_bounds__(256)
void softmax_kernel(const float* __restrict__ scores, const float* __restrict__ mask,
                    bf16_t* __restrict__ attn, float scale) {
    const int row  = blockIdx.x;
    const int head = blockIdx.y;
    const float* sp = scores + ((size_t)head * S_LEN + row) * S_LEN;
    const float* mp = mask + (size_t)row * S_LEN;
    bf16_t* dp = attn + ((size_t)head * S_LEN + row) * S_LEN;
    float vals[4];
    float lmax = -3.4e38f;
    int cnt = 0;
    for (int j = threadIdx.x; j < S_LEN; j += 256) {
        float v = sp[j] * scale + mp[j];
        vals[cnt++] = v;
        lmax = fmaxf(lmax, v);
    }
    __shared__ float red[256];
    red[threadIdx.x] = lmax; __syncthreads();
    for (int s = 128; s > 0; s >>= 1) {
        if (threadIdx.x < s) red[threadIdx.x] = fmaxf(red[threadIdx.x], red[threadIdx.x + s]);
        __syncthreads();
    }
    float m = red[0]; __syncthreads();
    float lsum = 0.f;
    for (int i = 0; i < cnt; ++i) { vals[i] = __expf(vals[i] - m); lsum += vals[i]; }
    red[threadIdx.x] = lsum; __syncthreads();
    for (int s = 128; s > 0; s >>= 1) {
        if (threadIdx.x < s) red[threadIdx.x] += red[threadIdx.x + s];
        __syncthreads();
    }
    float inv = 1.f / red[0];
    cnt = 0;
    for (int j = threadIdx.x; j < S_LEN; j += 256) dp[j] = (bf16_t)(vals[cnt++] * inv);
}

__global__ void swiglu_kernel(const float* __restrict__ g, const float* __restrict__ u,
                              bf16_t* __restrict__ o, long long n) {
    long long i = (long long)blockIdx.x * blockDim.x + threadIdx.x;
    long long stride = (long long)gridDim.x * blockDim.x;
    for (; i < n; i += stride) {
        float gv = g[i];
        o[i] = (bf16_t)((gv / (1.f + __expf(-gv))) * u[i]);
    }
}

// ---------------------------------------------------------------------------
// Host launcher
// ---------------------------------------------------------------------------
static inline void launch_gemm(const bf16_t* A, long long aB, int lda,
                               const bf16_t* B, long long bB, int bDiv, int ldb,
                               float* C, bf16_t* Cb, long long cB, int ldc,
                               const float* R, int ldr,
                               int M, int N, int K, int Z, hipStream_t stream) {
    if (N % 128 == 0) {
        dim3 grid(N / 128, M / BM, Z);
        gemm_bf16_kernel<128><<<grid, 256, 0, stream>>>(A, aB, lda, B, bB, bDiv, ldb,
                                                        C, Cb, cB, ldc, R, ldr, K);
    } else {
        dim3 grid(N / 64, M / BM, Z);
        gemm_bf16_kernel<64><<<grid, 256, 0, stream>>>(A, aB, lda, B, bB, bDiv, ldb,
                                                       C, Cb, cB, ldc, R, ldr, K);
    }
}

extern "C" void kernel_launch(void* const* d_in, const int* in_sizes, int n_in,
                              void* d_out, int out_size, void* d_ws, size_t ws_size,
                              hipStream_t stream) {
    (void)in_sizes; (void)n_in; (void)out_size; (void)ws_size;
    const float* x    = (const float*)d_in[0];
    const float* mask = (const float*)d_in[1];
    const float* cosc = (const float*)d_in[3];
    const float* sinc = (const float*)d_in[4];
    const float* ln1  = (const float*)d_in[5];
    const float* ln2  = (const float*)d_in[6];
    const float* Wq   = (const float*)d_in[7];
    const float* Wk   = (const float*)d_in[8];
    const float* Wv   = (const float*)d_in[9];
    const float* Wo   = (const float*)d_in[10];
    const float* Wg   = (const float*)d_in[11];
    const float* Wu   = (const float*)d_in[12];
    const float* Wd   = (const float*)d_in[13];

    // --- workspace carve-up (256B aligned) ---
    char* ws = (char*)d_ws;
    size_t off = 0;
    auto alloc = [&](size_t bytes) -> void* {
        void* p = ws + off;
        off += (bytes + 255) & ~(size_t)255;
        return p;
    };
    const size_t L = 2;
    bf16_t* wq_bf = (bf16_t*)alloc(L * HID * HID * 2);
    bf16_t* wk_bf = (bf16_t*)alloc(L * KVW * HID * 2);
    bf16_t* wv_bf = (bf16_t*)alloc(L * KVW * HID * 2);
    bf16_t* wo_bf = (bf16_t*)alloc(L * HID * HID * 2);
    bf16_t* wg_bf = (bf16_t*)alloc(L * (size_t)INTER_ * HID * 2);
    bf16_t* wu_bf = (bf16_t*)alloc(L * (size_t)INTER_ * HID * 2);
    bf16_t* wd_bf = (bf16_t*)alloc(L * (size_t)HID * INTER_ * 2);
    float*  hbuf  = (float*) alloc((size_t)S_LEN * HID * 4);
    bf16_t* hn_bf = (bf16_t*)alloc((size_t)S_LEN * HID * 2);
    float*  qf    = (float*) alloc((size_t)S_LEN * HID * 4);
    float*  kf    = (float*) alloc((size_t)S_LEN * KVW * 4);
    float*  vf    = (float*) alloc((size_t)S_LEN * KVW * 4);
    bf16_t* q_bf  = (bf16_t*)alloc((size_t)S_LEN * HID * 2);
    bf16_t* k_bf  = (bf16_t*)alloc((size_t)S_LEN * KVW * 2);
    bf16_t* vT_bf = (bf16_t*)alloc((size_t)S_LEN * KVW * 2);
    float*  scores= (float*) alloc((size_t)NH * S_LEN * S_LEN * 4);
    bf16_t* attn  = (bf16_t*)alloc((size_t)NH * S_LEN * S_LEN * 2);
    bf16_t* ao_bf = (bf16_t*)alloc((size_t)S_LEN * HID * 2);
    float*  gate  = (float*) alloc((size_t)S_LEN * INTER_ * 4);
    float*  up    = (float*) alloc((size_t)S_LEN * INTER_ * 4);
    bf16_t* t_bf  = (bf16_t*)alloc((size_t)S_LEN * INTER_ * 2);

    // --- output layout: h [S*HID], new_k [2][8][S][64], new_v [2][8][S][64]
    float* outH = (float*)d_out;
    float* outK = outH + (size_t)S_LEN * HID;
    float* outV = outK + L * (size_t)NKV * S_LEN * HD;

    // --- per-call weight conversion fp32 -> bf16 ---
    auto conv = [&](const float* src, bf16_t* dst, long long n) {
        f32_to_bf16_kernel<<<4096, 256, 0, stream>>>(src, dst, n);
    };
    conv(Wq, wq_bf, (long long)L * HID * HID);
    conv(Wk, wk_bf, (long long)L * KVW * HID);
    conv(Wv, wv_bf, (long long)L * KVW * HID);
    conv(Wo, wo_bf, (long long)L * HID * HID);
    conv(Wg, wg_bf, (long long)L * INTER_ * HID);
    conv(Wu, wu_bf, (long long)L * INTER_ * HID);
    conv(Wd, wd_bf, (long long)L * HID * INTER_);

    // h = x
    hipMemcpyAsync(hbuf, x, (size_t)S_LEN * HID * 4, hipMemcpyDeviceToDevice, stream);

    for (int i = 0; i < (int)L; ++i) {
        const bf16_t* wq = wq_bf + (size_t)i * HID * HID;
        const bf16_t* wk = wk_bf + (size_t)i * KVW * HID;
        const bf16_t* wv = wv_bf + (size_t)i * KVW * HID;
        const bf16_t* wo = wo_bf + (size_t)i * HID * HID;
        const bf16_t* wg = wg_bf + (size_t)i * (size_t)INTER_ * HID;
        const bf16_t* wu = wu_bf + (size_t)i * (size_t)INTER_ * HID;
        const bf16_t* wd = wd_bf + (size_t)i * (size_t)HID * INTER_;

        // 1) hn = rmsnorm(h, ln1) -> bf16
        rmsnorm_kernel<<<S_LEN, 256, 0, stream>>>(hbuf, ln1 + (size_t)i * HID, hn_bf, HID);

        // 2) q/k/v projections
        launch_gemm(hn_bf, 0, HID, wq, 0, 1, HID, qf, nullptr, 0, HID, nullptr, 0,
                    S_LEN, HID, HID, 1, stream);
        launch_gemm(hn_bf, 0, HID, wk, 0, 1, HID, kf, nullptr, 0, KVW, nullptr, 0,
                    S_LEN, KVW, HID, 1, stream);
        launch_gemm(hn_bf, 0, HID, wv, 0, 1, HID, vf, nullptr, 0, KVW, nullptr, 0,
                    S_LEN, KVW, HID, 1, stream);

        // 3) RoPE + cache outputs + bf16 packs
        rope_q_kernel<<<(S_LEN * HID + 255) / 256, 256, 0, stream>>>(qf, cosc, sinc, q_bf);
        rope_kv_kernel<<<(S_LEN * KVW + 255) / 256, 256, 0, stream>>>(
            kf, vf, cosc, sinc, k_bf, vT_bf,
            outK + (size_t)i * NKV * S_LEN * HD,
            outV + (size_t)i * NKV * S_LEN * HD);

        // 4) scores[h] = q_h @ k_{h/4}^T  (32 heads, GQA bDiv=4)
        launch_gemm(q_bf, HD, HID, k_bf, HD, 4, KVW,
                    scores, nullptr, (long long)S_LEN * S_LEN, S_LEN, nullptr, 0,
                    S_LEN, S_LEN, HD, NH, stream);

        // 5) softmax(scale + mask) -> attn bf16
        {
            dim3 g(S_LEN, NH, 1);
            softmax_kernel<<<g, 256, 0, stream>>>(scores, mask, attn, 0.125f);
        }

        // 6) out[h] = attn_h @ v_{h/4}; writes bf16 directly into merged [S,2048]
        launch_gemm(attn, (long long)S_LEN * S_LEN, S_LEN,
                    vT_bf, (long long)HD * S_LEN, 4, S_LEN,
                    nullptr, ao_bf, HD, HID, nullptr, 0,
                    S_LEN, HD, S_LEN, NH, stream);

        // 7) h = h + out @ Wo^T
        launch_gemm(ao_bf, 0, HID, wo, 0, 1, HID, hbuf, nullptr, 0, HID, hbuf, HID,
                    S_LEN, HID, HID, 1, stream);

        // 8) MLP
        rmsnorm_kernel<<<S_LEN, 256, 0, stream>>>(hbuf, ln2 + (size_t)i * HID, hn_bf, HID);
        launch_gemm(hn_bf, 0, HID, wg, 0, 1, HID, gate, nullptr, 0, INTER_, nullptr, 0,
                    S_LEN, INTER_, HID, 1, stream);
        launch_gemm(hn_bf, 0, HID, wu, 0, 1, HID, up, nullptr, 0, INTER_, nullptr, 0,
                    S_LEN, INTER_, HID, 1, stream);
        swiglu_kernel<<<4096, 256, 0, stream>>>(gate, up, t_bf, (long long)S_LEN * INTER_);
        launch_gemm(t_bf, 0, INTER_, wd, 0, 1, INTER_, hbuf, nullptr, 0, HID, hbuf, HID,
                    S_LEN, HID, INTER_, 1, stream);
    }

    // final hidden state -> d_out
    hipMemcpyAsync(outH, hbuf, (size_t)S_LEN * HID * 4, hipMemcpyDeviceToDevice, stream);
}
